// Wrapped8BitLinear_86019605004765
// MI455X (gfx1250) — matrix operand order repaired and checked by reading, not compiled, not run on hardware
//
#include <hip/hip_runtime.h>

typedef __attribute__((ext_vector_type(16))) _Float16 v16h;
typedef __attribute__((ext_vector_type(8)))  _Float16 v8h;
typedef __attribute__((ext_vector_type(4)))  _Float16 v4h;
typedef __attribute__((ext_vector_type(8)))  float    v8f;
typedef __attribute__((ext_vector_type(4)))  int      v4i;
typedef __attribute__((ext_vector_type(4)))  unsigned int u32x4;
typedef __attribute__((ext_vector_type(8)))  int      i32x8;
typedef __attribute__((ext_vector_type(4)))  int      i32x4;

#define DIN      4096
#define DOUT     4096
#define M_TOTAL  8192

#define BM 128
#define BN 128
#define BK 64
#define LDSS 72            // BK + 8 halves pad -> 144B row stride (16B aligned)
#define NSTAGE (DIN / BK)  // 64

#define USE_TDM __has_builtin(__builtin_amdgcn_tensor_load_to_lds)

__global__ __launch_bounds__(256)
void int8linear_wmma_f16(const _Float16* __restrict__ x,
                         const int*      __restrict__ CB,
                         const _Float16* __restrict__ SCB,
                         const _Float16* __restrict__ bias,
                         _Float16*       __restrict__ out)
{
    __shared__ _Float16 Al[2][BM * LDSS];   // x tiles (double buffered)
    __shared__ _Float16 Bl[2][BN * LDSS];   // dequantized W tiles
    __shared__ float    sScale[BN];         // SCB/127 for this N block

    const int tid  = threadIdx.x;
    const int lane = tid & 31;
    const int wave = tid >> 5;
    const int mBlock = blockIdx.y * BM;
    const int nBlock = blockIdx.x * BN;
    const int waveM = (wave >> 2) * 64;     // 2 waves along M
    const int waveN = (wave & 3) * 32;      // 4 waves along N
    const int lrow  = lane & 15;
    const int lhalf = (lane >> 4) & 1;

    if (tid < BN) sScale[tid] = (float)SCB[nBlock + tid] * (1.0f / 127.0f);

#if USE_TDM
    // ---- static part of the TDM descriptor (D#) for the x tile ----
    // 2D fp16 tensor [M_TOTAL, DIN]; tile [BM rows, BK cols]; LDS padding:
    // 16B (pad_amount=3 -> 4 DWORDs) every 128B (pad_interval=4 -> 32 DWORDs)
    // reproduces the 72-half row stride in LDS.
    i32x8 g1;
    g1[0] = (1 << 16)           // data_size = 2 bytes
          | (1 << 20)           // pad_enable
          | (4 << 22)           // pad_interval: 32 DWORDs = 128B
          | (3 << 25);          // pad_amount:   4 DWORDs = 16B
    g1[1] = (DIN & 0xFFFF) << 16;                 // tensor_dim0[15:0]
    g1[2] = ((M_TOTAL & 0xFFFF) << 16);           // dim0 hi (0) | tensor_dim1 lo
    g1[3] = (BK << 16);                           // dim1 hi (0) | tile_dim0
    g1[4] = BM;                                   // tile_dim1 | tile_dim2(0)
    g1[5] = DIN;                                  // tensor_dim0_stride lo
    g1[6] = 0;
    g1[7] = 0;
    const i32x4 gz = {0, 0, 0, 0};
#if __clang_major__ >= 23
    const i32x8 gz8 = {0, 0, 0, 0, 0, 0, 0, 0};
#endif
    unsigned aBase[2];
    aBase[0] = (unsigned)(unsigned long long)
               (__attribute__((address_space(3))) _Float16*)&Al[0][0];
    aBase[1] = (unsigned)(unsigned long long)
               (__attribute__((address_space(3))) _Float16*)&Al[1][0];
    const unsigned long long gArow =
        (unsigned long long)(x + (size_t)mBlock * DIN);
#endif

    // ---- stage helpers ----
    auto issueA = [&](int k0, int buf) {
#if USE_TDM
        if (wave == 0) {
            unsigned long long ga = gArow + (unsigned long long)k0 * 2ull;
            u32x4 g0;
            g0.x = 1u;                                        // count=1
            g0.y = aBase[buf];                                // lds_addr
            g0.z = (unsigned)ga;                              // global_addr lo
            g0.w = (unsigned)((ga >> 32) & 0x01FFFFFFull)     // global_addr hi
                 | (2u << 30);                                // type=2 (image)
#if __clang_major__ >= 23
            __builtin_amdgcn_tensor_load_to_lds(g0, g1, gz, gz, gz8, 0);
#else
            __builtin_amdgcn_tensor_load_to_lds(g0, g1, gz, gz, 0);
#endif
        }
#else
        #pragma unroll
        for (int i = 0; i < 4; ++i) {
            int idx = tid + i * 256;
            int row = idx >> 3;
            int c8  = idx & 7;
            v4i d = *(const v4i*)(x + (size_t)(mBlock + row) * DIN + k0 + c8 * 8);
            *(v4i*)(&Al[buf][row * LDSS + c8 * 8]) = d;
        }
#endif
    };

    v4i bq[8];
    auto loadB = [&](int k0) {
        #pragma unroll
        for (int i = 0; i < 8; ++i) {
            int idx = tid + i * 256;
            int row = idx >> 4;
            int c4  = idx & 15;
            bq[i] = *(const v4i*)(CB + (size_t)(nBlock + row) * DIN + k0 + c4 * 4);
        }
    };
    auto storeB = [&](int buf) {
        #pragma unroll
        for (int i = 0; i < 8; ++i) {
            int idx = tid + i * 256;
            int row = idx >> 4;
            int c4  = idx & 15;
            float s = sScale[row];
            v4h h;
            h.x = (_Float16)((float)bq[i].x * s);
            h.y = (_Float16)((float)bq[i].y * s);
            h.z = (_Float16)((float)bq[i].z * s);
            h.w = (_Float16)((float)bq[i].w * s);
            *(v4h*)(&Bl[buf][row * LDSS + c4 * 4]) = h;
        }
    };

    v8f acc[4][2];
    #pragma unroll
    for (int mt = 0; mt < 4; ++mt)
        #pragma unroll
        for (int nt = 0; nt < 2; ++nt)
            #pragma unroll
            for (int r = 0; r < 8; ++r)
                acc[mt][nt][r] = 0.0f;

    // ---- prologue: stage 0 into buffer 0 ----
    issueA(0, 0);
    loadB(0);
    __syncthreads();            // sScale visible
    storeB(0);
#if USE_TDM
    if (wave == 0) __builtin_amdgcn_s_wait_tensorcnt(0);
#endif
    __syncthreads();            // buffer 0 ready

    // ---- main pipeline: one barrier per stage ----
    for (int s = 0; s < NSTAGE; ++s) {
        const int cur  = s & 1;
        const int nxt  = cur ^ 1;
        const bool more = (s + 1) < NSTAGE;

        if (more) {
            issueA((s + 1) * BK, nxt);   // TDM streams next x tile into LDS
            loadB((s + 1) * BK);         // next CB tile into registers
        }

        const _Float16* Ac = &Al[cur][0];
        const _Float16* Bc = &Bl[cur][0];
        #pragma unroll
        for (int ks = 0; ks < BK; ks += 32) {
            v16h afrag[4];
            #pragma unroll
            for (int mt = 0; mt < 4; ++mt) {
                const _Float16* base =
                    &Ac[(waveM + mt * 16 + lrow) * LDSS + ks + lhalf * 8];
                v8h lo = *(const v8h*)(base);
                v8h hi = *(const v8h*)(base + 16);
                afrag[mt] = __builtin_shufflevector(lo, hi,
                    0,1,2,3,4,5,6,7,8,9,10,11,12,13,14,15);
            }
            v16h bfrag[2];
            #pragma unroll
            for (int nt = 0; nt < 2; ++nt) {
                const _Float16* base =
                    &Bc[(waveN + nt * 16 + lrow) * LDSS + ks + lhalf * 8];
                v8h lo = *(const v8h*)(base);
                v8h hi = *(const v8h*)(base + 16);
                bfrag[nt] = __builtin_shufflevector(lo, hi,
                    0,1,2,3,4,5,6,7,8,9,10,11,12,13,14,15);
            }
            #pragma unroll
            for (int mt = 0; mt < 4; ++mt)
                #pragma unroll
                for (int nt = 0; nt < 2; ++nt)
                    acc[mt][nt] = __builtin_amdgcn_wmma_f32_16x16x32_f16(
                        false, afrag[mt], false, bfrag[nt],
                        (short)0, acc[mt][nt], false, false);
        }

        if (more) {
            storeB(nxt);                 // dequantized W tile -> other buffer
#if USE_TDM
            if (wave == 0) __builtin_amdgcn_s_wait_tensorcnt(0);
#endif
        }
        __syncthreads();
    }

    // ---- epilogue: bias + non-temporal fp16 store ----
    #pragma unroll
    for (int nt = 0; nt < 2; ++nt) {
        int nCol = nBlock + waveN + nt * 16 + lrow;
        float bv = (float)bias[nCol];
        #pragma unroll
        for (int mt = 0; mt < 4; ++mt) {
            int mBase = mBlock + waveM + mt * 16 + lhalf * 8;
            #pragma unroll
            for (int r = 0; r < 8; ++r) {
                __builtin_nontemporal_store(
                    (_Float16)(acc[mt][nt][r] + bv),
                    &out[(size_t)(mBase + r) * DOUT + nCol]);
            }
        }
    }
}

extern "C" void kernel_launch(void* const* d_in, const int* in_sizes, int n_in,
                              void* d_out, int out_size, void* d_ws, size_t ws_size,
                              hipStream_t stream) {
    const _Float16* x    = (const _Float16*)d_in[0];
    const int*      CB   = (const int*)d_in[1];
    const _Float16* SCB  = (const _Float16*)d_in[2];
    const _Float16* bias = (const _Float16*)d_in[3];
    _Float16*       out  = (_Float16*)d_out;

    dim3 grid(DOUT / BN, M_TOTAL / BM);   // 32 x 64 = 2048 workgroups
    dim3 block(256);
    int8linear_wmma_f16<<<grid, block, 0, stream>>>(x, CB, SCB, bias, out);
}
